// MultiHeadAttention_50697793962302
// MI455X (gfx1250) — matrix operand order repaired and checked by reading, not compiled, not run on hardware
//
#include <hip/hip_runtime.h>

#define B_    8
#define S_    1024
#define H_    16
#define DH_   64
#define DEMB_ 1024

typedef __attribute__((ext_vector_type(16))) __bf16 v16bf;
typedef __attribute__((ext_vector_type(8)))  __bf16 v8bf;
typedef __attribute__((ext_vector_type(8)))  float  v8f;
typedef __attribute__((ext_vector_type(4)))  float  v4f;

static __device__ inline v8f zero8() {
    v8f z;
#pragma unroll
    for (int i = 0; i < 8; ++i) z[i] = 0.f;
    return z;
}

static __device__ inline v8f wmma_bf16(v16bf a, v16bf b, v8f c) {
    return __builtin_amdgcn_wmma_f32_16x16x32_bf16(false, a, false, b, (short)0, c,
                                                   false, false);
}

static __device__ inline float fast_exp2(float x) {
#if __has_builtin(__builtin_amdgcn_exp2f)
    return __builtin_amdgcn_exp2f(x);
#else
    return exp2f(x);
#endif
}

// Assemble a 16-bit A-layout fragment from two contiguous 8-half loads.
static __device__ inline v16bf afrag(const __bf16* lo, const __bf16* hi) {
    v8bf a0 = *(const v8bf*)lo;
    v8bf a1 = *(const v8bf*)hi;
    v16bf r;
#pragma unroll
    for (int e = 0; e < 8; ++e) { r[e] = a0[e]; r[8 + e] = a1[e]; }
    return r;
}

// ---------------------------------------------------------------------------
// Kernel 1: per-head QKV projection, fp32 -> bf16 fragments -> WMMA -> bf16 ws
//   Q, K stored row-major [bh][s][64] (transposed via LDS: packed 16B DS
//   writes in d-major, coalesced b128 global stores in s-major).
//   V stored transposed [bh][d][s] with packed 16B stores.
//   1/sqrt(DH) folded into Q.  Each wave: 16 rows of one head, 24 WMMA.
// ---------------------------------------------------------------------------
#define STRIDE_ 24   // LDS stage row stride in halfs: 48B, 16B-aligned, conflict-free

__global__ __launch_bounds__(256, 2) void qkv_proj_k(
    const float* __restrict__ x,
    const float* __restrict__ Wq, const float* __restrict__ bq,
    const float* __restrict__ Wk, const float* __restrict__ bk,
    const float* __restrict__ Wv, const float* __restrict__ bv,
    __bf16* __restrict__ Qw, __bf16* __restrict__ Kw, __bf16* __restrict__ Vt)
{
    // per-wave d-major stage tile: [d=64][m=16 (+pad)]
    __shared__ __attribute__((aligned(16))) __bf16 stage[8][64][STRIDE_];

    const int tile = blockIdx.x;          // 64 tiles of 128 rows over B*S
    const int h    = blockIdx.y;          // head
    const int wv   = threadIdx.x >> 5;    // wave id (8 waves)
    const int lane = threadIdx.x & 31;
    const int l16  = lane & 15;
    const int hf   = lane >> 4;

    const int row0 = tile * 128 + wv * 16;    // row = b*S + s
    const int b    = row0 / S_;
    const int sb   = row0 - b * S_;
    const int bh   = b * H_ + h;

    // ---- A fragments of X (16 rows x 64 d, two 32-K chunks), reused for Q/K/V
    v16bf A[2];
    {
        const float* xr = x + (size_t)(row0 + l16) * DEMB_ + h * DH_;
#pragma unroll
        for (int c = 0; c < 2; ++c) {
            const float* p0 = xr + c * 32 + hf * 16;      // elems [0..7]  : k = hf*16 + e
            const float* p1 = p0 + 8;                     // elems [8..15] : k = hf*16 + 8 + e
            v4f a0 = *(const v4f*)p0, a1 = *(const v4f*)(p0 + 4);
            v4f b0 = *(const v4f*)p1, b1 = *(const v4f*)(p1 + 4);
#pragma unroll
            for (int e = 0; e < 4; ++e) {
                A[c][e]      = (__bf16)a0[e];
                A[c][4 + e]  = (__bf16)a1[e];
                A[c][8 + e]  = (__bf16)b0[e];
                A[c][12 + e] = (__bf16)b1[e];
            }
        }
    }

    const float* Wmat[3] = { Wq + h * DH_ * DH_, Wk + h * DH_ * DH_, Wv + h * DH_ * DH_ };
    const float* bias[3] = { bq + h * DH_, bk + h * DH_, bv + h * DH_ };

#pragma unroll
    for (int p = 0; p < 3; ++p) {
        v8f accs[4];
#pragma unroll
        for (int j = 0; j < 4; ++j) {                      // output n-tile (16 cols of e)
            v8f acc = zero8();
#pragma unroll
            for (int c = 0; c < 2; ++c) {                  // contraction chunk (32 of d)
                // B fragment: B[k][n] = W[e=n][d=k]; per lane 16 consecutive floats
                const float* wp = Wmat[p] + (size_t)(j * 16 + l16) * DH_ + c * 32 + hf * 16;
                v4f w0 = *(const v4f*)wp,       w1 = *(const v4f*)(wp + 4);
                v4f w2 = *(const v4f*)(wp + 8), w3 = *(const v4f*)(wp + 12);
                v16bf bf;
#pragma unroll
                for (int e = 0; e < 4; ++e) {
                    bf[e]      = (__bf16)w0[e];
                    bf[4 + e]  = (__bf16)w1[e];
                    bf[8 + e]  = (__bf16)w2[e];
                    bf[12 + e] = (__bf16)w3[e];
                }
                acc = wmma_bf16(A[c], bf, acc);
            }
            const float bb = bias[p][j * 16 + l16];
            const float sc = (p == 0) ? 0.125f : 1.0f;     // fold 1/sqrt(64) into Q
#pragma unroll
            for (int v = 0; v < 8; ++v) acc[v] = (acc[v] + bb) * sc;
            accs[j] = acc;
        }

        if (p < 2) {
            // ---- transpose via LDS: packed d-major writes (16B each) ...
            __bf16* st = &stage[wv][0][0];
#pragma unroll
            for (int j = 0; j < 4; ++j) {
                v8bf pk;
#pragma unroll
                for (int v = 0; v < 8; ++v) pk[v] = (__bf16)accs[j][v];
                *(v8bf*)(st + (size_t)(j * 16 + l16) * STRIDE_ + 8 * hf) = pk;
            }
            __asm__ volatile("s_wait_dscnt 0" ::: "memory");  // cross-lane DS RAW
            // ---- ... then coalesced s-major readback: 4x b128 stores per lane-group
            __bf16* gp = (p == 0 ? Qw : Kw) + ((size_t)bh * S_ + sb) * DH_;
#pragma unroll
            for (int i = 0; i < 4; ++i) {
                const int m  = i * 4 + (lane >> 3);        // output row (s)
                const int d0 = (lane & 7) * 8;             // 8 consecutive d
                v8bf pk;
#pragma unroll
                for (int jj = 0; jj < 8; ++jj) pk[jj] = st[(size_t)(d0 + jj) * STRIDE_ + m];
                *(v8bf*)(gp + (size_t)m * DH_ + d0) = pk;
            }
            __asm__ volatile("" ::: "memory");             // order vs next p's DS writes
        } else {
            // ---- V transposed: 8 consecutive s per lane -> packed 16B stores
#pragma unroll
            for (int j = 0; j < 4; ++j) {
                v8bf pk;
#pragma unroll
                for (int v = 0; v < 8; ++v) pk[v] = (__bf16)accs[j][v];
                *(v8bf*)(Vt + ((size_t)bh * DH_ + j * 16 + l16) * S_ + sb + 8 * hf) = pk;
            }
        }
    }
}

// ---------------------------------------------------------------------------
// Kernel 2: flash attention in transposed orientation.
//   S^T = K . Q^T  (A = K tile, B = Q^T, loop-invariant)
//   O^T = V^T . P^T (A = V^T tile, B = P^T via one half-wave exchange)
// Keys live in the VGPR dim, q in the lane dim: softmax reduction is an
// in-register tree + a single shfl_xor(16); no LDS at all.
// ---------------------------------------------------------------------------
__global__ __launch_bounds__(256, 2) void flash_attn_k(
    const __bf16* __restrict__ Qw, const __bf16* __restrict__ Kw,
    const __bf16* __restrict__ Vt, float* __restrict__ out)
{
    const int qt   = blockIdx.x;          // 0..7 q-tiles of 128
    const int bh   = blockIdx.y;          // 0..127
    const int b    = bh >> 4;
    const int h    = bh & (H_ - 1);
    const int wv   = threadIdx.x >> 5;
    const int lane = threadIdx.x & 31;
    const int l16  = lane & 15;
    const int hf   = lane >> 4;
    const int q0   = qt * 128 + wv * 16;

    // ---- Q^T as B-operand fragments (n = q = l16): one 32B load per chunk
    v16bf QB[2];
    {
        const __bf16* qr = Qw + ((size_t)bh * S_ + q0 + l16) * DH_;
#pragma unroll
        for (int c = 0; c < 2; ++c)
            QB[c] = *(const v16bf*)(qr + c * 32 + hf * 16);
    }

    float mrun = -1e30f, lrun = 0.f;      // per-lane (= per q column) stats
    v8f   acc[4];                         // O^T: d = j2*16 + v + 8*hf, q = l16
#pragma unroll
    for (int j = 0; j < 4; ++j) acc[j] = zero8();

    const __bf16* Kb = Kw + (size_t)bh * S_ * DH_;
    const __bf16* Vb = Vt + (size_t)bh * DH_ * S_;

    for (int kb = 0; kb < S_; kb += 32) {
        if (kb + 32 < S_) {               // pull next K/V strip into WGP$
            __builtin_prefetch(Kb + (size_t)(kb + 32 + lane) * DH_, 0, 3);
            __builtin_prefetch(Vb + (size_t)(lane * 2) * S_ + kb + 32, 0, 3);
            __builtin_prefetch(Vb + (size_t)(lane * 2 + 1) * S_ + kb + 32, 0, 3);
        }

        // ---- S^T[32k x 16q]: key = kb + 16j + v + 8*hf, q = l16
        v8f s0 = zero8(), s1 = zero8();
#pragma unroll
        for (int c = 0; c < 2; ++c) {
            const __bf16* kr = Kb + (size_t)(kb + l16) * DH_ + c * 32;
            s0 = wmma_bf16(afrag(kr + hf * 16, kr + hf * 16 + 8), QB[c], s0);
        }
#pragma unroll
        for (int c = 0; c < 2; ++c) {
            const __bf16* kr = Kb + (size_t)(kb + 16 + l16) * DH_ + c * 32;
            s1 = wmma_bf16(afrag(kr + hf * 16, kr + hf * 16 + 8), QB[c], s1);
        }

        // ---- online softmax over the 32 keys (VGPR tree + 1 shuffle each)
        v8f t;
#pragma unroll
        for (int v = 0; v < 8; ++v) t[v] = fmaxf(s0[v], s1[v]);
        float mx = fmaxf(fmaxf(fmaxf(t[0], t[1]), fmaxf(t[2], t[3])),
                         fmaxf(fmaxf(t[4], t[5]), fmaxf(t[6], t[7])));
        mx = fmaxf(mx, __shfl_xor(mx, 16));
        const float mn = fmaxf(mrun, mx);
#pragma unroll
        for (int v = 0; v < 8; ++v) {
            s0[v] = fast_exp2((s0[v] - mn) * 1.44269504f);
            s1[v] = fast_exp2((s1[v] - mn) * 1.44269504f);
        }
#pragma unroll
        for (int v = 0; v < 8; ++v) t[v] = s0[v] + s1[v];
        float rs = (((t[0] + t[1]) + (t[2] + t[3])) + ((t[4] + t[5]) + (t[6] + t[7])));
        rs += __shfl_xor(rs, 16);
        const float f = fast_exp2((mrun - mn) * 1.44269504f);
        lrun = lrun * f + rs;
        mrun = mn;
#pragma unroll
        for (int j = 0; j < 4; ++j)
#pragma unroll
            for (int v = 0; v < 8; ++v) acc[j][v] *= f;

        // ---- P^T C-layout -> B-layout: one half-wave exchange, no LDS.
        //  B element e (0..15) needs key hf*16+e; own half holds 8, partner 8.
        v16bf PB;
#pragma unroll
        for (int v = 0; v < 8; ++v) {
            const float send = hf ? s0[v] : s1[v];
            const float recv = __shfl_xor(send, 16);
            const float lo   = hf ? recv : s0[v];     // keys hf*16 + v
            const float hi   = hf ? s1[v] : recv;     // keys hf*16 + 8 + v
            PB[v]     = (__bf16)lo;
            PB[8 + v] = (__bf16)hi;
        }

        // ---- O^T += V^T . P^T  (A = V^T tile: contiguous key runs per lane)
#pragma unroll
        for (int j2 = 0; j2 < 4; ++j2) {
            const __bf16* vr = Vb + (size_t)(j2 * 16 + l16) * S_ + kb;
            acc[j2] = wmma_bf16(afrag(vr + hf * 16, vr + hf * 16 + 8), PB, acc[j2]);
        }
    }

    // ---- finalize: d is consecutive per lane -> 32B vector stores
    const float inv = 1.f / lrun;
    float* op = out + ((size_t)b * S_ + q0 + l16) * DEMB_ + h * DH_ + 8 * hf;
#pragma unroll
    for (int j2 = 0; j2 < 4; ++j2) {
        v8f o;
#pragma unroll
        for (int v = 0; v < 8; ++v) o[v] = acc[j2][v] * inv;
        *(v8f*)(op + j2 * 16) = o;
    }
}

// ---------------------------------------------------------------------------
extern "C" void kernel_launch(void* const* d_in, const int* in_sizes, int n_in,
                              void* d_out, int out_size, void* d_ws, size_t ws_size,
                              hipStream_t stream)
{
    (void)in_sizes; (void)n_in; (void)out_size; (void)ws_size;
    const float* x  = (const float*)d_in[0];
    const float* Wq = (const float*)d_in[1];
    const float* bq = (const float*)d_in[2];
    const float* Wk = (const float*)d_in[3];
    const float* bk = (const float*)d_in[4];
    const float* Wv = (const float*)d_in[5];
    const float* bv = (const float*)d_in[6];

    const size_t n  = (size_t)B_ * H_ * S_ * DH_;   // 8M elems, 16 MB each in bf16
    __bf16* Qw = (__bf16*)d_ws;
    __bf16* Kw = Qw + n;
    __bf16* Vt = Kw + n;                            // 48 MB workspace total

    qkv_proj_k<<<dim3((B_ * S_) / 128, H_), 256, 0, stream>>>(
        x, Wq, bq, Wk, bk, Wv, bv, Qw, Kw, Vt);
    flash_attn_k<<<dim3(S_ / 128, B_ * H_), 256, 0, stream>>>(
        Qw, Kw, Vt, (float*)d_out);
}
